// GWRNN_4183298146631
// MI455X (gfx1250) — matrix operand order repaired and checked by reading, not compiled, not run on hardware
//
#include <hip/hip_runtime.h>
#include <hip/hip_bf16.h>
#include <math.h>

#define H      1024
#define ISZ    256
#define OSZ    128
#define BATCH  64
#define TSTEPS 512
#define ALPHA  0.1f
#define NWG    16
#define CHUNK  64              // H / NWG rows of W per workgroup
#define LDH    1032            // padded LDS row stride (u16) for K=1024 (bank skew 4)
#define LDI    264             // padded LDS row stride (u16) for K=256

typedef __attribute__((ext_vector_type(16))) __bf16 v16bf;
typedef __attribute__((ext_vector_type(8)))  float  v8f;
typedef __attribute__((ext_vector_type(4)))  float  v4f;
typedef unsigned short u16t;
typedef __attribute__((ext_vector_type(8))) u16t v8u16;
typedef __attribute__((ext_vector_type(4))) u16t v4u16;

union FragU { v16bf bf; v8u16 h[2]; u16t us[16]; };

__device__ __forceinline__ u16t f2bf(float f) {
    unsigned u = __builtin_bit_cast(unsigned, f);
    u += 0x7fffu + ((u >> 16) & 1u);           // round-to-nearest-even
    return (u16t)(u >> 16);
}

__device__ __forceinline__ float htanh(float f) {
#if __has_builtin(__builtin_amdgcn_tanhf)
    return __builtin_amdgcn_tanhf(f);          // gfx1250 v_tanh_f32
#else
    return tanhf(f);
#endif
}

// Async global->LDS 16B copy (GV mode, per-lane addresses), tracked by ASYNCcnt.
__device__ __forceinline__ void async_g2l_b128(void* ldsp, const void* gp) {
    unsigned           lds = (unsigned)(size_t)ldsp;            // addr[31:0] = LDS offset
    unsigned long long ga  = (unsigned long long)(size_t)gp;
    asm volatile("global_load_async_to_lds_b128 %0, %1, off"
                 :: "v"(lds), "v"(ga) : "memory");
}
__device__ __forceinline__ void wait_asynccnt0() {
#if __has_builtin(__builtin_amdgcn_s_wait_asynccnt)
    __builtin_amdgcn_s_wait_asynccnt(0);
#else
    asm volatile("s_wait_asynccnt 0x0" ::: "memory");
#endif
}

// A fragment (16x32 bf16): per-lane K map = {k0+8*hi+0..7, k0+16+8*hi+0..7}
__device__ __forceinline__ v16bf load_fragA(const u16t* base, int row, int ld,
                                            int k0, int hi) {
    const u16t* p = base + (size_t)row * ld + k0 + (hi ? 8 : 0);
    FragU f;
    f.h[0] = *(const v8u16*)(p);
    f.h[1] = *(const v8u16*)(p + 16);
    return f.bf;
}

// B fragment (32x16 bf16): per-lane K map = {k0+8*hi+0..7, k0+16+8*hi+0..7}
__device__ __forceinline__ v16bf load_fragB(const u16t* base, int row, int ld,
                                            int k0, int hi) {
    const u16t* p = base + (size_t)row * ld + k0 + (hi ? 8 : 0);
    FragU f;
    f.h[0] = *(const v8u16*)(p);
    f.h[1] = *(const v8u16*)(p + 16);
    return f.bf;
}

__device__ __forceinline__ v8f wmma_bf16(v16bf a, v16bf b, v8f c) {
    return __builtin_amdgcn_wmma_f32_16x16x32_bf16(false, a, false, b,
                                                   (short)0, c, false, false);
}

// ---------------- init: zero h ping-pong buffer + barrier counter ----------------
__global__ void init_kernel(unsigned* hz, unsigned* syncc) {
    int idx = blockIdx.x * blockDim.x + threadIdx.x;   // launched with exactly 65536
    hz[idx] = 0u;                                      // 2*64*1024 bf16 = 65536 dwords
    if (idx < 32) syncc[idx] = 0u;
}

// -------- prep: W_L = BB - M BB^T M^-1 (M diagonal) + bf16 weight conversion -----
__global__ void prep_kernel(const float* __restrict__ Bp, const float* __restrict__ Bm,
                            const float* __restrict__ Mh, const float* __restrict__ Whh,
                            const float* __restrict__ Wih, const float* __restrict__ Who,
                            u16t* __restrict__ wHH, u16t* __restrict__ wL,
                            u16t* __restrict__ wIH, u16t* __restrict__ wHO) {
    int idx = blockIdx.x * blockDim.x + threadIdx.x;
    int stride = gridDim.x * blockDim.x;
    for (int t = idx; t < H * H; t += stride) {
        int i = t >> 10, j = t & (H - 1);
        float bb  = Bp[t] * Bm[t];
        int   tji = j * H + i;
        float bbT = Bp[tji] * Bm[tji];
        float wl  = bb - Mh[i * H + i] * bbT / Mh[j * H + j];
        wL[t]  = f2bf(wl);
        wHH[t] = f2bf(Whh[t]);
        if (t < H * ISZ)  wIH[t] = f2bf(Wih[t]);
        if (t < OSZ * H)  wHO[t] = f2bf(Who[t]);
    }
}

// -------- prep: convert x to bf16 once (removes f2bf from the recurrent loop) ----
__global__ void xconv_kernel(const float* __restrict__ x, u16t* __restrict__ xbf) {
    const int n4 = (BATCH * TSTEPS * ISZ) / 4;
    int idx = blockIdx.x * blockDim.x + threadIdx.x;
    int stride = gridDim.x * blockDim.x;
    for (int i = idx; i < n4; i += stride) {
        v4f v = *(const v4f*)(x + (size_t)4 * i);
        v4u16 r;
        r[0] = f2bf(v[0]); r[1] = f2bf(v[1]); r[2] = f2bf(v[2]); r[3] = f2bf(v[3]);
        *(v4u16*)(xbf + (size_t)4 * i) = r;
    }
}

// ---------------- persistent recurrent kernel ----------------
__launch_bounds__(256, 1)
__global__ void rnn_kernel(const u16t* __restrict__ xbf,
                           const float* __restrict__ b_ih, const float* __restrict__ b_hh,
                           const float* __restrict__ b_o,
                           const u16t* __restrict__ wHHg, const u16t* __restrict__ wLg,
                           const u16t* __restrict__ wIHg, const u16t* __restrict__ wHOg,
                           u16t* __restrict__ hbf, unsigned* __restrict__ syncc,
                           float* __restrict__ out) {
    extern __shared__ u16t smem[];
    u16t* sWhh = smem;                          // CHUNK x LDH
    u16t* sWL  = smem + CHUNK * LDH;            // CHUNK x LDH
    u16t* sWih = smem + 2 * CHUNK * LDH;        // CHUNK x LDI

    const int wg        = blockIdx.x;
    const int tid       = threadIdx.x;
    const int chunkBase = wg * CHUNK;

    // --- resident weight slices into LDS via async DMA (16B units, skewed rows) ---
    for (int c = tid; c < CHUNK * (H / 8); c += blockDim.x) {
        int row = c / (H / 8), col8 = (c % (H / 8)) * 8;
        async_g2l_b128(sWhh + row * LDH + col8,
                       wHHg + (size_t)(chunkBase + row) * H + col8);
        async_g2l_b128(sWL + row * LDH + col8,
                       wLg + (size_t)(chunkBase + row) * H + col8);
    }
    for (int c = tid; c < CHUNK * (ISZ / 8); c += blockDim.x) {
        int row = c / (ISZ / 8), col8 = (c % (ISZ / 8)) * 8;
        async_g2l_b128(sWih + row * LDI + col8,
                       wIHg + (size_t)(chunkBase + row) * ISZ + col8);
    }
    wait_asynccnt0();
    __syncthreads();

    const int wave = tid >> 5;
    const int lane = tid & 31;
    const int lm   = lane & 15;   // M (A/C) or N (B/C) within 16x16 tile
    const int hi   = lane >> 4;   // half selector
    const int mt   = wave & 3;    // batch-tile of this wave (4 tiles cover B=64)
    const int ntp  = wave >> 2;   // this wave owns N-tiles {2*ntp, 2*ntp+1}

    const int   brow  = mt * 16 + lm;                        // batch row for A frags
    const u16t* xrow0 = xbf + (size_t)brow * TSTEPS * ISZ;

    float bias[2]; int coln[2];
#pragma unroll
    for (int e = 0; e < 2; ++e) {
        int col = chunkBase + (ntp * 2 + e) * 16 + lm;       // hidden column (fixed/lane)
        coln[e] = col;
        bias[e] = b_ih[col] + b_hh[col];
    }

    // output-projection duty: WGs 0..7, waves 0..3
    const bool doY   = (wg < (OSZ / 16)) && (wave < 4);
    const int  ocol  = wg * 16 + lm;
    const int  ymt   = wave;
    float      ybias = (wg < (OSZ / 16)) ? b_o[ocol] : 0.f;

    const v8f zro = {0.f, 0.f, 0.f, 0.f, 0.f, 0.f, 0.f, 0.f};
    v8f hreg[2] = {zro, zro};                                // persistent f32 state tiles

#pragma unroll 1
    for (int t = 0; t < TSTEPS; ++t) {
        const u16t* hin  = hbf + (size_t)(t & 1) * (BATCH * H);
        u16t*       hout = hbf + (size_t)((t + 1) & 1) * (BATCH * H);

        v8f accP[2] = {zro, zro};   // pre-activation (x*Wih^T + h*Whh^T)
        v8f accL[2] = {zro, zro};   // h*W_L^T

        // input projection, K = 256 (A from preconverted bf16 x)
        const u16t* xr = xrow0 + (size_t)t * ISZ;
        __builtin_prefetch(xr + ISZ, 0, 1);                  // next step's x row
#pragma unroll
        for (int k0 = 0; k0 < ISZ; k0 += 32) {
            v16bf a = load_fragA(xr, 0, ISZ, k0, hi);
#pragma unroll
            for (int e = 0; e < 2; ++e) {
                v16bf b = load_fragB(sWih, (ntp * 2 + e) * 16 + lm, LDI, k0, hi);
                accP[e] = wmma_bf16(a, b, accP[e]);
            }
        }
        // recurrent terms, K = 1024 (A from L2, B from LDS)
#pragma unroll 2
        for (int k0 = 0; k0 < H; k0 += 32) {
            v16bf a = load_fragA(hin, brow, H, k0, hi);
#pragma unroll
            for (int e = 0; e < 2; ++e) {
                int   nrow = (ntp * 2 + e) * 16 + lm;
                v16bf bh   = load_fragB(sWhh, nrow, LDH, k0, hi);
                v16bf bl   = load_fragB(sWL,  nrow, LDH, k0, hi);
                accP[e] = wmma_bf16(a, bh, accP[e]);
                accL[e] = wmma_bf16(a, bl, accL[e]);
            }
        }
        // fused pointwise update in f32; publish bf16 shadow state
#pragma unroll
        for (int e = 0; e < 2; ++e) {
#pragma unroll
            for (int vg = 0; vg < 8; ++vg) {
                float hold = hreg[e][vg];
                float pre  = accP[e][vg] + bias[e];
                float v    = hold + ALPHA * (-hold + htanh(pre) + accL[e][vg]);
                hreg[e][vg] = v;
                int b = mt * 16 + vg + 8 * hi;               // C/D layout: M = vg + 8*hi
                hout[(size_t)b * H + coln[e]] = f2bf(v);
            }
        }

        // ---- device-wide barrier (monotonic counter) ----
        __threadfence();
        __syncthreads();
        if (tid == 0) {
            atomicAdd(syncc, 1u);
            unsigned target = (unsigned)(t + 1) * NWG;
            while (__hip_atomic_load(syncc, __ATOMIC_ACQUIRE,
                                     __HIP_MEMORY_SCOPE_AGENT) < target)
                __builtin_amdgcn_s_sleep(2);
            __threadfence();
        }
        __syncthreads();

        // ---- y_t = h_{t+1} @ W_ho^T + b_o from freshly published state ----
        if (doY) {
            v8f accY = zro;
            const int yrow = ymt * 16 + lm;
#pragma unroll 2
            for (int k0 = 0; k0 < H; k0 += 32) {
                v16bf a = load_fragA(hout, yrow, H, k0, hi);
                v16bf b = load_fragB(wHOg, ocol, H, k0, hi);
                accY = wmma_bf16(a, b, accY);
            }
#pragma unroll
            for (int vg = 0; vg < 8; ++vg) {
                int b = ymt * 16 + vg + 8 * hi;
                out[((size_t)b * TSTEPS + t) * OSZ + ocol] = accY[vg] + ybias;
            }
        }
    }
}

extern "C" void kernel_launch(void* const* d_in, const int* in_sizes, int n_in,
                              void* d_out, int out_size, void* d_ws, size_t ws_size,
                              hipStream_t stream) {
    const float* x    = (const float*)d_in[0];
    const float* W_ih = (const float*)d_in[1];
    const float* b_ih = (const float*)d_in[2];
    const float* W_hh = (const float*)d_in[3];
    const float* b_hh = (const float*)d_in[4];
    const float* W_ho = (const float*)d_in[5];
    const float* b_o  = (const float*)d_in[6];
    const float* Bp   = (const float*)d_in[7];
    const float* Mh   = (const float*)d_in[8];
    const float* Bm   = (const float*)d_in[9];

    char* ws = (char*)d_ws;
    u16t*     hbf   = (u16t*)(ws);                 // 2 * 64*1024 bf16 = 256 KB
    u16t*     wHH   = (u16t*)(ws + 262144);        // 2 MB
    u16t*     wL    = (u16t*)(ws + 2359296);       // 2 MB
    u16t*     wIH   = (u16t*)(ws + 4456448);       // 512 KB
    u16t*     wHO   = (u16t*)(ws + 4980736);       // 256 KB
    unsigned* syncc = (unsigned*)(ws + 5242880);   // barrier counter (128 B)
    u16t*     xbf   = (u16t*)(ws + 5243008);       // 16 MB bf16 copy of x

    init_kernel<<<256, 256, 0, stream>>>((unsigned*)hbf, syncc);
    prep_kernel<<<1024, 256, 0, stream>>>(Bp, Bm, Mh, W_hh, W_ih, W_ho,
                                          wHH, wL, wIH, wHO);
    xconv_kernel<<<2048, 256, 0, stream>>>(x, xbf);
    size_t smem = (size_t)(2 * CHUNK * LDH + CHUNK * LDI) * sizeof(u16t); // 297,984 B
    rnn_kernel<<<NWG, 256, smem, stream>>>(xbf, b_ih, b_hh, b_o,
                                           wHH, wL, wIH, wHO, hbf, syncc,
                                           (float*)d_out);
}
